// FullSelfAttention_32710470927092
// MI455X (gfx1250) — matrix operand order repaired and checked by reading, not compiled, not run on hardware
//
#include <hip/hip_runtime.h>
#include <math.h>

// ---------------------------------------------------------------------------
// Ragged transformer block for MI455X (gfx1250, wave32, WMMA).
//   D=192, H=8, DH=24, DFF=384, 32 segments of 384+8i tokens, TOTAL=16256.
// All matmuls use v_wmma_f32_16x16x32_f16 (f16 A/B, f32 accumulate).
// Weights pre-converted to f16 (B fragments load directly as halfs).
// Attention fragment loads are unconditional + branch-free masking
// (no per-element EXEC churn); head-dim padding 24->32 done via cndmask.
// ---------------------------------------------------------------------------

typedef __attribute__((ext_vector_type(16))) _Float16 v16h;
typedef __attribute__((ext_vector_type(8)))  _Float16 v8h;
typedef __attribute__((ext_vector_type(4)))  _Float16 v4h;
typedef __attribute__((ext_vector_type(8)))  float    v8f;

#define TOTAL_TOK 16256
#define DMODEL    192
#define NHEAD     8
#define DHEAD     24
#define QKVLD     576
#define SC_STRIDE 640          // LDS score row stride (max n = 632)

static __device__ __forceinline__ v8f wmma_f16(v16h a, v16h b, v8f c) {
    return __builtin_amdgcn_wmma_f32_16x16x32_f16(false, a, false, b,
                                                  (short)0, c, false, false);
}
static __device__ __forceinline__ float fast_rcp(float x)  { return __builtin_amdgcn_rcpf(x); }
static __device__ __forceinline__ float fast_rsq(float x)  { return __builtin_amdgcn_rsqf(x); }
static __device__ __forceinline__ float fast_sigmoid(float z) {
    return fast_rcp(1.f + __expf(-z));
}

// A fragment: 16(M) x 32(K) f16 from a row-major f32 matrix (all K valid).
// lane<16: row=lane,    elems 0-7 -> K k0+0..7,  elems 8-15 -> K k0+16..23
// lane>=16: row=lane-16, elems 0-7 -> K k0+8..15, elems 8-15 -> K k0+24..31
static __device__ __forceinline__ v16h load_a_frag(const float* __restrict__ rowp,
                                                   int k0, int lane) {
    const int klo = (lane < 16) ? 0 : 8;
    const float4* p0 = (const float4*)(rowp + k0 + klo);
    const float4* p1 = (const float4*)(rowp + k0 + klo + 16);
    float4 x0 = p0[0], x1 = p0[1], y0 = p1[0], y1 = p1[1];
    v16h a;
    a[0]=(_Float16)x0.x;  a[1]=(_Float16)x0.y;  a[2]=(_Float16)x0.z;  a[3]=(_Float16)x0.w;
    a[4]=(_Float16)x1.x;  a[5]=(_Float16)x1.y;  a[6]=(_Float16)x1.z;  a[7]=(_Float16)x1.w;
    a[8]=(_Float16)y0.x;  a[9]=(_Float16)y0.y;  a[10]=(_Float16)y0.z; a[11]=(_Float16)y0.w;
    a[12]=(_Float16)y1.x; a[13]=(_Float16)y1.y; a[14]=(_Float16)y1.z; a[15]=(_Float16)y1.w;
    return a;
}

// A fragment, contiguous K (elem e of lane half h -> d = 16h+e); K-dim is the 24-wide head slice padded to 32.
// All loads unconditional (reads stay inside the 576-float qkv row /
// workspace); elements with d>=24 are zeroed via value select.
static __device__ __forceinline__ v16h load_a_frag_pad24(const float* __restrict__ rowp,
                                                         int lane) {
    const int klo = (lane < 16) ? 0 : 8;
    const float4* p0 = (const float4*)(rowp + 2 * klo);
    const float4* p1 = (const float4*)(rowp + 2 * klo + 8);
    float4 x0 = p0[0], x1 = p0[1], y0 = p1[0], y1 = p1[1];
    v16h a;
    a[0]=(_Float16)x0.x;  a[1]=(_Float16)x0.y;  a[2]=(_Float16)x0.z;  a[3]=(_Float16)x0.w;
    a[4]=(_Float16)x1.x;  a[5]=(_Float16)x1.y;  a[6]=(_Float16)x1.z;  a[7]=(_Float16)x1.w;
    // elems 8..15 hold d = 2*klo+8..2*klo+15: valid for klo==0, all padding for klo==8
    const bool hi_ok = (klo == 0);
    a[8]  = hi_ok ? (_Float16)y0.x : (_Float16)0.f;
    a[9]  = hi_ok ? (_Float16)y0.y : (_Float16)0.f;
    a[10] = hi_ok ? (_Float16)y0.z : (_Float16)0.f;
    a[11] = hi_ok ? (_Float16)y0.w : (_Float16)0.f;
    a[12] = hi_ok ? (_Float16)y1.x : (_Float16)0.f;
    a[13] = hi_ok ? (_Float16)y1.y : (_Float16)0.f;
    a[14] = hi_ok ? (_Float16)y1.z : (_Float16)0.f;
    a[15] = hi_ok ? (_Float16)y1.w : (_Float16)0.f;
    return a;
}

// B fragment with K-dim = head slice padded 24->32: lane-half kbhf in {0,16};
// element e -> d = kbhf+e.  Unconditional 4x b128, zero d>=24 via select.
static __device__ __forceinline__ v16h load_b_frag_pad24(const float* __restrict__ rowp,
                                                         int kbhf) {
    const float4* p = (const float4*)(rowp + kbhf);
    float4 x0 = p[0], x1 = p[1], x2 = p[2], x3 = p[3];
    const bool hi_ok = (kbhf == 0);        // elems 8..15 -> d = kbhf+8..15
    v16h b;
    b[0]=(_Float16)x0.x;  b[1]=(_Float16)x0.y;  b[2]=(_Float16)x0.z;  b[3]=(_Float16)x0.w;
    b[4]=(_Float16)x1.x;  b[5]=(_Float16)x1.y;  b[6]=(_Float16)x1.z;  b[7]=(_Float16)x1.w;
    b[8]  = hi_ok ? (_Float16)x2.x : (_Float16)0.f;
    b[9]  = hi_ok ? (_Float16)x2.y : (_Float16)0.f;
    b[10] = hi_ok ? (_Float16)x2.z : (_Float16)0.f;
    b[11] = hi_ok ? (_Float16)x2.w : (_Float16)0.f;
    b[12] = hi_ok ? (_Float16)x3.x : (_Float16)0.f;
    b[13] = hi_ok ? (_Float16)x3.y : (_Float16)0.f;
    b[14] = hi_ok ? (_Float16)x3.z : (_Float16)0.f;
    b[15] = hi_ok ? (_Float16)x3.w : (_Float16)0.f;
    return b;
}

// B fragment: 32(K) x 16(N) f16 where B[k][n] = Wh[n][k], Wh row-major [N,K] f16.
static __device__ __forceinline__ v16h load_b_frag_h(const _Float16* __restrict__ Wh,
                                                     int ldw, int n, int k0, int lane) {
    const v8h* q = (const v8h*)(Wh + (size_t)n * ldw + k0 + ((lane < 16) ? 0 : 8));
    v8h lo = q[0], hi = q[2];
    return __builtin_shufflevector(lo, hi, 0,1,2,3,4,5,6,7,8,9,10,11,12,13,14,15);
}

// ---------------------------------------------------------------------------
// Weight f32 -> f16 conversion (vectorized by 4).
// ---------------------------------------------------------------------------
__global__ __launch_bounds__(256)
void cvt_w_kernel(const float* __restrict__ src, _Float16* __restrict__ dst, int n4) {
    int i = blockIdx.x * 256 + threadIdx.x;
    if (i < n4) {
        float4 v = ((const float4*)src)[i];
        v4h h;
        h[0] = (_Float16)v.x; h[1] = (_Float16)v.y;
        h[2] = (_Float16)v.z; h[3] = (_Float16)v.w;
        ((v4h*)dst)[i] = h;
    }
}

// ---------------------------------------------------------------------------
// C[M,N] = act( (ACCUM ? C : 0) + A[M,K] @ Wh[N,K]^T + bias[N] )
// One 16(M) x 64(N) output block per wave.  M mult 16, N mult 64, K mult 32.
// ---------------------------------------------------------------------------
template <int K, int ACCUM, int ACT>
__global__ __launch_bounds__(128)
void gemm_wmma_kernel(const float* __restrict__ A, int lda,
                      const _Float16* __restrict__ Wh, int ldw,
                      const float* __restrict__ bias,
                      float* __restrict__ C, int ldc, int M, int N) {
    const int lane = threadIdx.x & 31;
    const int wave = blockIdx.x * (blockDim.x >> 5) + (threadIdx.x >> 5);
    const int tiles_n = N >> 6, tiles_m = M >> 4;
    if (wave >= tiles_m * tiles_n) return;            // wave-uniform exit
    const int m0 = (wave / tiles_n) << 4;
    const int n0 = (wave % tiles_n) << 6;

    const float* arow = A + (size_t)(m0 + (lane & 15)) * lda;
    const int nc = n0 + (lane & 15);

    v8f acc0 = {}, acc1 = {}, acc2 = {}, acc3 = {};
#pragma unroll
    for (int k0 = 0; k0 < K; k0 += 32) {
        if (k0 + 32 < K) __builtin_prefetch(arow + k0 + 32, 0, 0);
        v16h a  = load_a_frag(arow, k0, lane);
        v16h b0 = load_b_frag_h(Wh, ldw, nc,      k0, lane);
        v16h b1 = load_b_frag_h(Wh, ldw, nc + 16, k0, lane);
        v16h b2 = load_b_frag_h(Wh, ldw, nc + 32, k0, lane);
        v16h b3 = load_b_frag_h(Wh, ldw, nc + 48, k0, lane);
        acc0 = wmma_f16(a, b0, acc0);
        acc1 = wmma_f16(a, b1, acc1);
        acc2 = wmma_f16(a, b2, acc2);
        acc3 = wmma_f16(a, b3, acc3);
    }

    const int rbase = (lane < 16) ? 0 : 8;
    auto epi = [&](v8f acc, int col) {
        float bv = bias ? bias[col] : 0.f;
        for (int r = 0; r < 8; ++r) {
            size_t idx = (size_t)(m0 + rbase + r) * ldc + col;
            float v = acc[r];
            if (ACCUM) v += C[idx];
            v += bv;
            if (ACT) v = v * fast_rcp(1.f + __expf(-v));   // SiLU
            C[idx] = v;
        }
    };
    epi(acc0, nc);
    epi(acc1, nc + 16);
    epi(acc2, nc + 32);
    epi(acc3, nc + 48);
}

// ---------------------------------------------------------------------------
// Ragged multi-head attention.  One wave per (segment, 16-query block, head).
// qkv layout: [TOTAL, 576] = [Q(192) | K(192) | V(192)], head h owns cols
// h*24..h*24+23 of each sub-block.  Output: attn_out [TOTAL, 192].
// Two-pass softmax with a 16 x 640 f32 score strip in LDS (40 KB).
// ---------------------------------------------------------------------------
__global__ __launch_bounds__(32)
void attn_kernel(const float* __restrict__ qkv, float* __restrict__ out) {
    __shared__ float sc[16 * SC_STRIDE];
    const int lane = threadIdx.x & 31;
    const int head = blockIdx.x & 7;
    int rem = blockIdx.x >> 3;

    // locate (segment, query block) — sizes are compile-time static
    int n = 384, q0 = 0, seg0 = 0;
    for (int b = 0; b < 32; ++b) {
        int nb  = 384 + 8 * b;
        int nqb = (nb + 15) >> 4;
        if (rem < nqb) { n = nb; q0 = rem << 4; seg0 = 384 * b + 4 * b * (b - 1); break; }
        rem -= nqb;
    }

    const float* Qp = qkv +       head * DHEAD;
    const float* Kp = qkv + 192 + head * DHEAD;
    const float* Vp = qkv + 384 + head * DHEAD;
    const float scale = 0.20412414523193154f;         // 1/sqrt(24)
    const int col  = lane & 15;
    const int klo  = (lane < 16) ? 0 : 8;             // A-frag K split
    const int kbhf = (lane < 16) ? 0 : 16;            // B-frag K split
    const int rb   = (lane < 16) ? 0 : 8;             // C/D row split

    // ---- loop-invariant A fragment: Q rows q0..q0+15, d padded 24->32 ----
    int qrow = q0 + col; if (qrow >= n) qrow = n - 1;
    const v16h aq = load_a_frag_pad24(Qp + (size_t)(seg0 + qrow) * QKVLD, lane);

    // ---- pass 1: S = Q K^T / sqrt(dh) -> LDS ----
    for (int kb0 = 0; kb0 < n; kb0 += 16) {
        int key = kb0 + col;
        int keyc = (key < n) ? key : n - 1;
        v16h bk = load_b_frag_pad24(Kp + (size_t)(seg0 + keyc) * QKVLD, kbhf);
        v8f z = {};
        v8f s = wmma_f16(aq, bk, z);
        for (int r = 0; r < 8; ++r)
            sc[(rb + r) * SC_STRIDE + key] = (key < n) ? s[r] * scale : -1e30f;
    }
    __syncthreads();

    // ---- pass 2: row softmax (lane pair l / l+16 shares row l) ----
    {
        const int row = lane & 15;
        const int half = lane >> 4;
        float mx = -1e30f;
        for (int k = half; k < n; k += 2) mx = fmaxf(mx, sc[row * SC_STRIDE + k]);
        mx = fmaxf(mx, __shfl_xor(mx, 16, 32));
        float sum = 0.f;
        for (int k = half; k < n; k += 2) sum += __expf(sc[row * SC_STRIDE + k] - mx);
        sum += __shfl_xor(sum, 16, 32);
        float inv = fast_rcp(sum);
        for (int k = half; k < n; k += 2)
            sc[row * SC_STRIDE + k] = __expf(sc[row * SC_STRIDE + k] - mx) * inv;
    }
    // zero the never-written tail columns [ceil16(n), ceil32(n)) so pass-3
    // P-fragment loads can be unconditional
    {
        const int n16 = (n + 15) & ~15;
        const int n32t = (n + 31) & ~31;
        for (int k = n16 + lane; k < n32t; k += 32)
            for (int row = 0; row < 16; ++row) sc[row * SC_STRIDE + k] = 0.f;
    }
    __syncthreads();

    // ---- pass 3: O = P V  (two d-tiles: 0..15 and 16..23+pad) ----
    v8f o0 = {}, o1 = {};
    const int n32 = (n + 31) & ~31;
    const int d0 = col, d1 = 16 + col;
    const bool d1ok = (col < 8);                      // d1 < 24
    for (int kb0 = 0; kb0 < n32; kb0 += 32) {
        v16h ap;
        for (int e = 0; e < 16; ++e) {
            int key = kb0 + 2 * klo + e;
            ap[e] = (_Float16)sc[col * SC_STRIDE + key];     // tail is exact 0
        }
        v16h bv0, bv1;
        for (int e = 0; e < 16; ++e) {
            int key = kb0 + kbhf + e;
            bool ok = (key < n);
            const float* vp = Vp + (size_t)(seg0 + (ok ? key : n - 1)) * QKVLD;
            float v0 = vp[d0];                                // unconditional
            float v1 = vp[d1];                                // unconditional
            bv0[e] = ok ? (_Float16)v0 : (_Float16)0.f;
            bv1[e] = (ok && d1ok) ? (_Float16)v1 : (_Float16)0.f;
        }
        o0 = wmma_f16(ap, bv0, o0);
        o1 = wmma_f16(ap, bv1, o1);
    }

    for (int r = 0; r < 8; ++r) {
        int q = q0 + rb + r;
        if (q < n) {
            float* op = out + (size_t)(seg0 + q) * DMODEL + head * DHEAD;
            op[d0] = o0[r];
            if (d1ok) op[d1] = o1[r];
        }
    }
}

// ---------------------------------------------------------------------------
// h = LN1( g*attn_out + (1-g)*x ),  g = sigmoid(glog).  One wave per token.
// ---------------------------------------------------------------------------
__global__ __launch_bounds__(256)
void gate_ln_kernel(const float* __restrict__ of, const float* __restrict__ x,
                    const float* __restrict__ glog,
                    const float* __restrict__ gamma, const float* __restrict__ beta,
                    float* __restrict__ h, int total) {
    const int token = blockIdx.x * 8 + (threadIdx.x >> 5);
    const int lane  = threadIdx.x & 31;
    if (token >= total) return;
    const float* ofp = of   + (size_t)token * DMODEL;
    const float* xp  = x    + (size_t)token * DMODEL;
    const float* gp  = glog + (size_t)token * DMODEL;
    float* hp        = h    + (size_t)token * DMODEL;

    float vals[6], s = 0.f, s2 = 0.f;
    for (int i = 0; i < 6; ++i) {
        int c = lane + 32 * i;
        float g  = fast_sigmoid(gp[c]);
        float hv = g * ofp[c] + (1.f - g) * xp[c];
        vals[i] = hv; s += hv; s2 += hv * hv;
    }
    for (int m = 16; m >= 1; m >>= 1) { s += __shfl_xor(s, m, 32); s2 += __shfl_xor(s2, m, 32); }
    float mean = s * (1.f / DMODEL);
    float var  = s2 * (1.f / DMODEL) - mean * mean;
    float rs   = fast_rsq(var + 1e-5f);
    for (int i = 0; i < 6; ++i) {
        int c = lane + 32 * i;
        hp[c] = (vals[i] - mean) * rs * gamma[c] + beta[c];
    }
}

// out = LN2(h + y)
__global__ __launch_bounds__(256)
void ln2_kernel(const float* __restrict__ h, const float* __restrict__ y,
                const float* __restrict__ gamma, const float* __restrict__ beta,
                float* __restrict__ out, int total) {
    const int token = blockIdx.x * 8 + (threadIdx.x >> 5);
    const int lane  = threadIdx.x & 31;
    if (token >= total) return;
    const float* hp = h + (size_t)token * DMODEL;
    const float* yp = y + (size_t)token * DMODEL;
    float* op       = out + (size_t)token * DMODEL;

    float vals[6], s = 0.f, s2 = 0.f;
    for (int i = 0; i < 6; ++i) {
        int c = lane + 32 * i;
        float v = hp[c] + yp[c];
        vals[i] = v; s += v; s2 += v * v;
    }
    for (int m = 16; m >= 1; m >>= 1) { s += __shfl_xor(s, m, 32); s2 += __shfl_xor(s2, m, 32); }
    float mean = s * (1.f / DMODEL);
    float var  = s2 * (1.f / DMODEL) - mean * mean;
    float rs   = fast_rsq(var + 1e-5f);
    for (int i = 0; i < 6; ++i) {
        int c = lane + 32 * i;
        op[c] = (vals[i] - mean) * rs * gamma[c] + beta[c];
    }
}

// ---------------------------------------------------------------------------
extern "C" void kernel_launch(void* const* d_in, const int* in_sizes, int n_in,
                              void* d_out, int out_size, void* d_ws, size_t ws_size,
                              hipStream_t stream) {
    (void)in_sizes; (void)n_in; (void)out_size; (void)ws_size;
    const float* x    = (const float*)d_in[0];
    // d_in[1] (sizes) unused: segment lengths are static in the reference.
    const float* Wqkv = (const float*)d_in[2];
    const float* bqkv = (const float*)d_in[3];
    const float* Wo   = (const float*)d_in[4];
    const float* bo   = (const float*)d_in[5];
    const float* Wg   = (const float*)d_in[6];
    const float* bg   = (const float*)d_in[7];
    const float* g1   = (const float*)d_in[8];
    const float* be1  = (const float*)d_in[9];
    const float* W1   = (const float*)d_in[10];
    const float* b1   = (const float*)d_in[11];
    const float* W2   = (const float*)d_in[12];
    const float* b2   = (const float*)d_in[13];
    const float* g2   = (const float*)d_in[14];
    const float* be2  = (const float*)d_in[15];
    float* out = (float*)d_out;
    float* ws  = (float*)d_ws;

    const int T = TOTAL_TOK;
    // ---- workspace layout ----
    // f32 region (reused):  peak = T*960 floats (~62 MB)
    float* qkvb = ws;                          // [T,576] qkv; later h | ff1
    float* r1   = ws + (size_t)T * 576;        // [T,192] attn; later gate logits
    float* r2   = r1 + (size_t)T * 192;        // [T,192] out_flat; later y
    float* hb   = qkvb;                        // [T,192] h (qkv dead)
    float* ffb  = qkvb + (size_t)T * 192;      // [T,384] silu(ff1)
    // f16 weight region (~0.74 MB) after the f32 region
    _Float16* wh     = (_Float16*)(ws + (size_t)T * 960);
    _Float16* Wqkv16 = wh;                       // 576*192
    _Float16* Wo16   = Wqkv16 + 576 * 192;       // 192*192
    _Float16* Wg16   = Wo16   + 192 * 192;       // 192*384
    _Float16* W116   = Wg16   + 192 * 384;       // 384*192
    _Float16* W216   = W116   + 384 * 192;       // 192*384

    auto cvtw = [&](const float* src, _Float16* dst, int n) {
        int n4 = n >> 2;
        cvt_w_kernel<<<(n4 + 255) / 256, 256, 0, stream>>>(src, dst, n4);
    };
    cvtw(Wqkv, Wqkv16, 576 * 192);
    cvtw(Wo,   Wo16,   192 * 192);
    cvtw(Wg,   Wg16,   192 * 384);
    cvtw(W1,   W116,   384 * 192);
    cvtw(W2,   W216,   192 * 384);

    auto blocks_for = [&](int M, int N) {
        int tiles = (M >> 4) * (N >> 6);                      // 16x64 per wave
        return (tiles + 3) / 4;                               // 4 waves/block
    };

    // 1) qkv = x @ Wqkv^T + bqkv
    gemm_wmma_kernel<192, 0, 0><<<blocks_for(T, 576), 128, 0, stream>>>(
        x, 192, Wqkv16, 192, bqkv, qkvb, 576, T, 576);
    // 2) ragged masked multi-head attention (1024 query blocks x 8 heads)
    attn_kernel<<<1024 * 8, 32, 0, stream>>>(qkvb, r1);
    // 3) out_flat = attn @ Wo^T + bo
    gemm_wmma_kernel<192, 0, 0><<<blocks_for(T, 192), 128, 0, stream>>>(
        r1, 192, Wo16, 192, bo, r2, 192, T, 192);
    // 4) gate logits = out_flat @ Wg[:, :192]^T + x @ Wg[:, 192:]^T + bg
    gemm_wmma_kernel<192, 0, 0><<<blocks_for(T, 192), 128, 0, stream>>>(
        r2, 192, Wg16, 384, nullptr, r1, 192, T, 192);
    gemm_wmma_kernel<192, 1, 0><<<blocks_for(T, 192), 128, 0, stream>>>(
        x, 192, Wg16 + 192, 384, bg, r1, 192, T, 192);
    // 5) h = LN1(sigmoid(glog)*out_flat + (1-sigmoid)*x)
    gate_ln_kernel<<<T / 8, 256, 0, stream>>>(r2, x, r1, g1, be1, hb, T);
    // 6) ff = silu(h @ W1^T + b1)
    gemm_wmma_kernel<192, 0, 1><<<blocks_for(T, 384), 128, 0, stream>>>(
        hb, 192, W116, 192, b1, ffb, 384, T, 384);
    // 7) y = ff @ W2^T + b2
    gemm_wmma_kernel<384, 0, 0><<<blocks_for(T, 192), 128, 0, stream>>>(
        ffb, 384, W216, 384, b2, r2, 192, T, 192);
    // 8) out = LN2(h + y)
    ln2_kernel<<<T / 8, 256, 0, stream>>>(hb, r2, g2, be2, out, T);
}
